// PLKBlock_90615220011482
// MI455X (gfx1250) — matrix operand order repaired and checked by reading, not compiled, not run on hardware
//
#include <hip/hip_runtime.h>
#include <hip/hip_bf16.h>

// PLK block for MI455X (gfx1250, wave32).
// - Depthwise 17x17 conv on v_wmma_f32_16x16x32_f16 via precomputed Toeplitz
//   weight tiles in LDS (k = x+dx, 17 WMMAs per 16x16 output tile).
// - Pointwise 1x1 conv as WMMA GEMM with 64(d) x 128(p) blocks; A and B tiles
//   are raw f16 and staged with global_load_async_to_lds_b128 + s_wait_asynccnt.
//
// Workspace layout:
//   hmid : NTOT f16     (post BN1 + depthwise conv)         50.3 MB
//   hact : NTOT f32     (post pointwise + BN2 + GELU)      100.7 MB
//   pool : B*C f32, gate : B*C f32
//   wh   : C*C f16      (pointwise weights pre-converted)   73.7 KB

typedef __attribute__((ext_vector_type(16))) _Float16 v16h;
typedef __attribute__((ext_vector_type(8)))  _Float16 v8h;
typedef __attribute__((ext_vector_type(8)))  float    v8f;
typedef __attribute__((address_space(3)))    _Float16 lds_f16;

#define BATCH 8
#define CH    192
#define HH    128
#define WW    128
#define KW    17
#define PADW  8
#define HIDN  48
#define HWP   (HH*WW)
#define NTOT  (BATCH*CH*HH*WW)
#define BN_EPS 1e-5f

// ---------------------------------------------------------------- K0: zero
__global__ __launch_bounds__(256) void k_zero_pool(float* __restrict__ pool) {
    int i = blockIdx.x * 256 + threadIdx.x;
    if (i < BATCH * CH) pool[i] = 0.f;
}

// ------------------------------------------------ K0b: weights f32 -> f16
__global__ __launch_bounds__(256) void k_cvt_w(const float* __restrict__ wpw,
                                               _Float16* __restrict__ wh) {
    int i = blockIdx.x * 256 + threadIdx.x;
    if (i < CH * CH) wh[i] = (_Float16)wpw[i];
}

// ------------------------------------------------- K1: BN1 + depthwise conv
// grid = (H/16, C, B), block = 256 (8 waves). Each wave: one 16x16 out tile.
__global__ __launch_bounds__(256) void k_bn_dwconv(
    const float* __restrict__ x,  const float* __restrict__ flt,
    const float* __restrict__ g1, const float* __restrict__ b1,
    const float* __restrict__ m1, const float* __restrict__ v1,
    _Float16* __restrict__ hmid)
{
    __shared__ _Float16 sIn[32][160];     // rows y0-8..y0+23, cols -8..135
    __shared__ _Float16 sT[KW][32][16];   // Toeplitz tiles, one per dy

    const int b    = blockIdx.z;
    const int c    = blockIdx.y;
    const int y0   = blockIdx.x * 16;
    const int tid  = threadIdx.x;
    const int lane = tid & 31;
    const int wave = tid >> 5;
    const int x0   = wave * 16;

    const float sc = g1[c] / sqrtf(v1[c] + BN_EPS);
    const float sh = b1[c] - m1[c] * sc;

    const float* plane = x + (size_t)(b * CH + c) * HWP;
    const float* fltc  = flt + c * KW * KW;

    // Stage BN'd input tile as f16 (clamped load + select: no divergent loads).
    for (int i = tid; i < 32 * 144; i += 256) {
        int r  = i / 144;
        int cc = i - r * 144;
        int gy = y0 + r - PADW;
        int gx = cc - PADW;
        int gyc = gy < 0 ? 0 : (gy > HH - 1 ? HH - 1 : gy);
        int gxc = gx < 0 ? 0 : (gx > WW - 1 ? WW - 1 : gx);
        float v = plane[gyc * WW + gxc] * sc + sh;
        bool in = (gy == gyc) && (gx == gxc);
        sIn[r][cc] = (_Float16)(in ? v : 0.f);
    }
    // Stage Toeplitz tiles: T[dy][k][n] = w[dy, k-n] for 0 <= k-n < 17 else 0.
    for (int i = tid; i < KW * 32 * 16; i += 256) {
        int dy = i >> 9;
        int k  = (i >> 4) & 31;
        int n  = i & 15;
        int idx  = k - n;
        int idxc = idx < 0 ? 0 : (idx > KW - 1 ? KW - 1 : idx);
        float wv = fltc[dy * KW + idxc];          // unconditional (cached) load
        sT[dy][k][n] = (idx == idxc) ? (_Float16)wv : (_Float16)0.f;
    }
    __syncthreads();

    const int m  = lane & 15;               // output row within tile (A rows)
    const int kb = (lane < 16) ? 0 : 8;     // K sub-block per 16-bit A layout
    v8f acc = {};

#pragma unroll
    for (int dy = 0; dy < KW; ++dy) {
        const _Float16* rowp = &sIn[m + dy][x0];
        v8h lo = *(const v8h*)(rowp + kb);
        v8h hi = *(const v8h*)(rowp + kb + 16);
        v16h a = __builtin_shufflevector(lo, hi, 0, 1, 2, 3, 4, 5, 6, 7,
                                         8, 9, 10, 11, 12, 13, 14, 15);
        v16h bm;
#pragma unroll
        for (int s = 0; s < 16; ++s)
            bm[s] = sT[dy][(s < 8) ? (kb + s) : (kb + 8 + s)][m];
        acc = __builtin_amdgcn_wmma_f32_16x16x32_f16(
            false, a, false, bm, (short)0, acc, false, false);
    }

    _Float16* outp = hmid + (size_t)(b * CH + c) * HWP;
    const int n  = lane & 15;
    const int mb = (lane < 16) ? 0 : 8;
#pragma unroll
    for (int v = 0; v < 8; ++v)
        outp[(y0 + mb + v) * WW + (x0 + n)] = (_Float16)acc[v];
}

// --------------------------- K2: pointwise GEMM + BN2 + GELU + pool partials
// grid = (HW/128, C/64, B), block = 256 (8 waves). Block tile: 64 d x 128 p.
// Each wave owns one 16-p column and 4 d-subtiles (4 accumulators); one B
// fragment feeds 4 WMMAs. A and B tiles staged with async global->LDS copies.
__global__ __launch_bounds__(256) void k_pw_bn_gelu(
    const _Float16* __restrict__ hmid, const _Float16* __restrict__ wh,
    const float* __restrict__ g2,      const float* __restrict__ b2,
    const float* __restrict__ m2,      const float* __restrict__ v2,
    float* __restrict__ hact,          float* __restrict__ pool)
{
    __shared__ _Float16 sA[64][32];     // weight tile: 64 d x 32 c
    __shared__ _Float16 sB[32][144];    // activation tile: 32 c x 128 p (+pad)
    __shared__ float sScale[64], sShift[64], sPool[64];

    const int b    = blockIdx.z;
    const int d0   = blockIdx.y * 64;
    const int p0   = blockIdx.x * 128;
    const int tid  = threadIdx.x;
    const int lane = tid & 31;
    const int wave = tid >> 5;
    const int pw   = p0 + wave * 16;

    if (tid < 64) {
        float s = g2[d0 + tid] / sqrtf(v2[d0 + tid] + BN_EPS);
        sScale[tid] = s;
        sShift[tid] = b2[d0 + tid] - m2[d0 + tid] * s;
        sPool[tid]  = 0.f;
    }

    const int m  = lane & 15;
    const int kb = (lane < 16) ? 0 : 8;
    v8f acc0 = {}, acc1 = {}, acc2 = {}, acc3 = {};

    for (int kk = 0; kk < CH; kk += 32) {
        __syncthreads();
        // Stage A: raw f16 copy of 64x32 weight tile (one 16B chunk/thread).
        {
            int r  = tid >> 2;           // 0..63
            int cb = (tid & 3) * 8;      // half offset within 32-c row
            lds_f16* dstp = (lds_f16*)&sA[r][cb];
            const _Float16* srcp = wh + (d0 + r) * CH + kk + cb;
            asm volatile("global_load_async_to_lds_b128 %0, %1, off"
                         :: "v"(dstp), "v"(srcp) : "memory");
        }
        // Stage B: raw f16 copy of 32x128 activation tile (two chunks/thread).
        for (int ch = tid; ch < 32 * 16; ch += 256) {
            int r  = ch >> 4;            // c row within slab
            int cb = (ch & 15) * 8;      // half offset within row
            lds_f16* dstp = (lds_f16*)&sB[r][cb];
            const _Float16* srcp =
                hmid + (size_t)(b * CH + kk + r) * HWP + p0 + cb;
            asm volatile("global_load_async_to_lds_b128 %0, %1, off"
                         :: "v"(dstp), "v"(srcp) : "memory");
        }
        asm volatile("s_wait_asynccnt 0x0" ::: "memory");
        __syncthreads();

        v16h bm;
#pragma unroll
        for (int s = 0; s < 16; ++s)
            bm[s] = sB[(s < 8) ? (kb + s) : (kb + 8 + s)][wave * 16 + m];
#pragma unroll
        for (int dt = 0; dt < 4; ++dt) {
            v8h lo = *(const v8h*)(&sA[dt * 16 + m][kb]);
            v8h hi = *(const v8h*)(&sA[dt * 16 + m][kb + 16]);
            v16h a = __builtin_shufflevector(lo, hi, 0, 1, 2, 3, 4, 5, 6, 7,
                                             8, 9, 10, 11, 12, 13, 14, 15);
            v8f* accp = dt == 0 ? &acc0 : dt == 1 ? &acc1 : dt == 2 ? &acc2
                                                                    : &acc3;
            *accp = __builtin_amdgcn_wmma_f32_16x16x32_f16(
                false, a, false, bm, (short)0, *accp, false, false);
        }
    }
    __syncthreads();

    const int n  = lane & 15;
    const int mb = (lane < 16) ? 0 : 8;
#pragma unroll
    for (int dt = 0; dt < 4; ++dt) {
        v8f accv = dt == 0 ? acc0 : dt == 1 ? acc1 : dt == 2 ? acc2 : acc3;
#pragma unroll
        for (int v = 0; v < 8; ++v) {
            int dr    = dt * 16 + mb + v;
            float val = accv[v] * sScale[dr] + sShift[dr];
            float ge  = 0.5f * val * (1.f + erff(val * 0.70710678118f));
            hact[(size_t)(b * CH + d0 + dr) * HWP + pw + n] = ge;
            atomicAdd(&sPool[dr], ge);
        }
    }
    __syncthreads();
    if (tid < 64)
        atomicAdd(&pool[b * CH + d0 + tid], sPool[tid]);
}

// ------------------------------------------------------------ K3: SE gating
__global__ __launch_bounds__(512) void k_se(
    const float* __restrict__ pool, const float* __restrict__ w1,
    const float* __restrict__ bb1,  const float* __restrict__ w2,
    const float* __restrict__ bb2,  float* __restrict__ gate)
{
    __shared__ float sP[BATCH * CH];
    __shared__ float sA1[BATCH * HIDN];
    const int tid = threadIdx.x;

    for (int i = tid; i < BATCH * CH; i += 512)
        sP[i] = pool[i] * (1.f / (float)HWP);
    __syncthreads();

    for (int i = tid; i < BATCH * HIDN; i += 512) {
        int b = i / HIDN, h = i - b * HIDN;
        float s = bb1[h];
        for (int c = 0; c < CH; ++c) s += sP[b * CH + c] * w1[h * CH + c];
        sA1[i] = s > 0.f ? s : 0.f;
    }
    __syncthreads();

    for (int i = tid; i < BATCH * CH; i += 512) {
        int b = i / CH, d = i - b * CH;
        float s = bb2[d];
        for (int h = 0; h < HIDN; ++h) s += sA1[b * HIDN + h] * w2[d * HIDN + h];
        gate[i] = 1.f / (1.f + expf(-s));
    }
}

// ------------------------------------------------- K4: residual + gamma*gate
__global__ __launch_bounds__(256) void k_finalize(
    const float* __restrict__ x,    const float* __restrict__ hact,
    const float* __restrict__ gate, const float* __restrict__ gamma,
    float* __restrict__ out)
{
    size_t i4 = (size_t)blockIdx.x * 256 + threadIdx.x;
    if (i4 >= (size_t)NTOT / 4) return;
    size_t i  = i4 * 4;
    int   bc  = (int)(i / HWP);          // = b*CH + c (HWP % 4 == 0)
    float g   = gamma[0] * gate[bc];
    const float4 xv = ((const float4*)x)[i4];
    const float4 hv = ((const float4*)hact)[i4];
    float4 o;
    o.x = xv.x + g * hv.x;
    o.y = xv.y + g * hv.y;
    o.z = xv.z + g * hv.z;
    o.w = xv.w + g * hv.w;
    ((float4*)out)[i4] = o;
}

// ------------------------------------------------------------------- launch
extern "C" void kernel_launch(void* const* d_in, const int* in_sizes, int n_in,
                              void* d_out, int out_size, void* d_ws, size_t ws_size,
                              hipStream_t stream) {
    const float* x    = (const float*)d_in[0];
    const float* flt  = (const float*)d_in[1];
    const float* g1   = (const float*)d_in[2];
    const float* b1   = (const float*)d_in[3];
    const float* m1   = (const float*)d_in[4];
    const float* v1   = (const float*)d_in[5];
    const float* wpw  = (const float*)d_in[6];
    const float* g2   = (const float*)d_in[7];
    const float* b2   = (const float*)d_in[8];
    const float* m2   = (const float*)d_in[9];
    const float* v2   = (const float*)d_in[10];
    const float* sw1  = (const float*)d_in[11];
    const float* sb1  = (const float*)d_in[12];
    const float* sw2  = (const float*)d_in[13];
    const float* sb2  = (const float*)d_in[14];
    const float* gam  = (const float*)d_in[15];
    float* out = (float*)d_out;

    char* ws = (char*)d_ws;
    _Float16* hmid = (_Float16*)ws;                       // NTOT f16
    float*    hact = (float*)(ws + (size_t)NTOT * 2);     // NTOT f32
    float*    pool = hact + (size_t)NTOT;                 // B*C
    float*    gate = pool + BATCH * CH;                   // B*C
    _Float16* wh   = (_Float16*)(gate + BATCH * CH);      // C*C f16

    k_zero_pool<<<(BATCH * CH + 255) / 256, 256, 0, stream>>>(pool);
    k_cvt_w<<<(CH * CH + 255) / 256, 256, 0, stream>>>(wpw, wh);

    k_bn_dwconv<<<dim3(HH / 16, CH, BATCH), 256, 0, stream>>>(
        x, flt, g1, b1, m1, v1, hmid);

    k_pw_bn_gelu<<<dim3(HWP / 128, CH / 64, BATCH), 256, 0, stream>>>(
        hmid, wh, g2, b2, m2, v2, hact, pool);

    k_se<<<1, 512, 0, stream>>>(pool, sw1, sb1, sw2, sb2, gate);

    k_finalize<<<(NTOT / 4 + 255) / 256, 256, 0, stream>>>(
        x, hact, gate, gam, out);
}
